// SourceStateBlock_40922448396565
// MI455X (gfx1250) — compile-verified, matrix operand order checked by reading, not run on hardware
//
#include <hip/hip_runtime.h>
#include <hip/hip_bf16.h>

#define BB 8
#define TT 512
#define SS 64
#define DD 256
#define HH 1024

typedef __bf16 bf16;
typedef __attribute__((ext_vector_type(16))) __bf16 v16bf;
typedef __attribute__((ext_vector_type(8)))  float  v8f;
typedef __attribute__((ext_vector_type(4)))  int    v4i;

#if defined(__gfx1250__) && __has_builtin(__builtin_amdgcn_global_load_async_to_lds_b128) && \
    __has_builtin(__builtin_amdgcn_s_wait_asynccnt)
#define HAVE_ASYNC 1
typedef __attribute__((address_space(1))) v4i as1_v4i;  // global (printed as __device__)
typedef __attribute__((address_space(3))) v4i as3_v4i;  // LDS (printed as __shared__)
#else
#define HAVE_ASYNC 0
#endif

__device__ inline v8f vzero() {
  v8f z = {0.f, 0.f, 0.f, 0.f, 0.f, 0.f, 0.f, 0.f};
  return z;
}

// ---------------------------------------------------------------------------
// Fragment-swizzled layout: element (p, k) of an operand (p = M-row for A,
// p = N-col for B) lives at
//   slot*512 + (hh*16 + p%16)*16 + e,  slot = (p/16)*nkch + k/32
//   hh = (k%32 >> 3) & 1, g = ((k%32 & 16) ? 4 : 0) + ((k%32 >> 1) & 3),
//   e = 2*g + k%2
// so one lane's 16 bf16 fragment values are 32 contiguous bytes.
// ---------------------------------------------------------------------------
__device__ inline int swz(int p, int k, int nkch) {
  const int pt = p >> 4, pl = p & 15;
  const int kc = k >> 5, kl = k & 31;
  const int hh = (kl >> 3) & 1;
  const int g = ((kl & 16) ? 4 : 0) + ((kl >> 1) & 3);
  const int e = 2 * g + (kl & 1);
  return ((((pt * nkch + kc) << 5) + (hh << 4) + pl) << 4) + e;
}

__device__ inline v16bf frag_ld(const bf16* p, int tile, int kc, int nkch, int lane) {
  return *(const v16bf*)(p + ((((tile * nkch + kc) << 5) + lane) << 4));
}

__device__ inline v8f wmma_bf16(v16bf a, v16bf b, v8f c) {
  return __builtin_amdgcn_wmma_f32_16x16x32_bf16(false, a, false, b, (short)0, c, false, false);
}

// Raw copy of n16 x 16-byte units from (pre-swizzled bf16) global into LDS.
// Uses the CDNA5 async LDS-DMA path when available.
__device__ inline void stage_copy(bf16* lds, const bf16* __restrict__ g, int n16, int tid) {
#if HAVE_ASYNC
  for (int i = tid; i < n16; i += 256) {
    __builtin_amdgcn_global_load_async_to_lds_b128(
        (as1_v4i*)(g + (size_t)i * 8), (as3_v4i*)(lds + (size_t)i * 8), 0, 0);
  }
#else
  const int4* src = (const int4*)g;
  int4* dst = (int4*)lds;
  for (int i = tid; i < n16; i += 256) dst[i] = src[i];
#endif
}

__device__ inline void stage_fence() {
#if HAVE_ASYNC
  __builtin_amdgcn_s_wait_asynccnt(0);
#endif
  __syncthreads();
}

// ===================== K0: weight prep =====================
// Convert an fp32 matrix into bf16 fragment-swizzled chunks in global scratch.
// Element (k, n) is read from W[k*sk + n*sn]. Output chunk order: [nb][kc],
// each chunk = [Nblk/16 tiles][32 lanes][16] bf16 (contiguous, LDS-ready).
__global__ __launch_bounds__(256) void k_prep(const float* __restrict__ W,
                                              bf16* __restrict__ dst, int sk, int sn,
                                              int K, int Nblk, int nNB) {
  const int nkch = K >> 5;
  const int ntile = Nblk >> 4;
  const int slotsPerNB = (nkch * ntile) << 5;
  const int total = nNB * slotsPerNB;
  for (int s = blockIdx.x * blockDim.x + threadIdx.x; s < total;
       s += gridDim.x * blockDim.x) {
    const int nb = s / slotsPerNB, s2 = s - nb * slotsPerNB;
    const int l = s2 & 31, u = s2 >> 5;
    const int tile = u % ntile, kc = u / ntile;
    const int hh = l >> 4, nl = l & 15;
    const int n = nb * Nblk + tile * 16 + nl;
    v16bf v;
#pragma unroll
    for (int g = 0; g < 8; ++g) {
      const int kb = kc * 32 + ((g & 4) ? 16 : 0) + hh * 8 + (g & 3) * 2;
      v[2 * g] = (bf16)W[(size_t)kb * sk + (size_t)n * sn];
      v[2 * g + 1] = (bf16)W[(size_t)(kb + 1) * sk + (size_t)n * sn];
    }
    *(v16bf*)(dst + (size_t)s * 16) = v;
  }
}

// LayerNorm stats for one row (4 lanes per row, shfl reduce).
__device__ inline void ln_stats(const float* __restrict__ row, int q,
                                float& mu, float& rs) {
  float s = 0.f, s2 = 0.f;
#pragma unroll 8
  for (int i = 0; i < 64; ++i) {
    const float v = row[q * 64 + i];
    s += v;
    s2 += v * v;
  }
  s += __shfl_xor(s, 1);
  s += __shfl_xor(s, 2);
  s2 += __shfl_xor(s2, 1);
  s2 += __shfl_xor(s2, 2);
  mu = s * (1.f / 256.f);
  const float var = s2 * (1.f / 256.f) - mu * mu;
  rs = rsqrtf(var + 1e-5f);
}

// LN 64x256 tile -> A-swizzled bf16 LDS (packed u32 stores).
__device__ inline void ln_to_lds_A(bf16* h, const float* __restrict__ xg,
                                   const float* __restrict__ gamma,
                                   const float* __restrict__ beta, int tid) {
  const int r = tid >> 2, q = tid & 3;
  const float* row = xg + r * DD;
  float mu, rs;
  ln_stats(row, q, mu, rs);
  unsigned* h32 = (unsigned*)h;
#pragma unroll 4
  for (int i = 0; i < 64; i += 2) {
    const int d = q * 64 + i;
    const float a0 = (row[d] - mu) * rs * gamma[d] + beta[d];
    const float a1 = (row[d + 1] - mu) * rs * gamma[d + 1] + beta[d + 1];
    union { bf16 b[2]; unsigned u; } pk;
    pk.b[0] = (bf16)a0;
    pk.b[1] = (bf16)a1;
    h32[swz(r, d, 8) >> 1] = pk.u;
  }
}

// LN 64x256 tile -> B-swizzled bf16 LDS (h used as B in adj @ h; K = 64 rows).
__device__ inline void ln_to_lds_B(bf16* h, const float* __restrict__ xg,
                                   const float* __restrict__ gamma,
                                   const float* __restrict__ beta, int tid) {
  const int r = tid >> 2, q = tid & 3;
  const float* row = xg + r * DD;
  float mu, rs;
  ln_stats(row, q, mu, rs);
#pragma unroll 4
  for (int i = 0; i < 64; ++i) {
    const int d = q * 64 + i;
    h[swz(d, r, 2)] = (bf16)((row[d] - mu) * rs * gamma[d] + beta[d]);
  }
}

__device__ inline float gelu_tanh(float x) {
  const float k0 = 0.7978845608028654f;  // sqrt(2/pi)
  const float x3 = x * x * x;
  return 0.5f * x * (1.f + tanhf(k0 * (x + 0.044715f * x3)));
}

// ===================== K1: graph mix =====================
// x1 = x + (adj @ LN(x)) @ g_w + g_b    (per (b,t) tile of 64x256)
__global__ __launch_bounds__(256) void k_graphmix(
    const float* __restrict__ x, const bf16* __restrict__ adjP,
    const float* __restrict__ gamma, const float* __restrict__ beta,
    const bf16* __restrict__ gwP, const float* __restrict__ gb,
    float* __restrict__ x1) {
  __shared__ alignas(32) bf16 hbuf[SS * DD];  // h (B-swz, K=64) then mixed (A-swz, K=256)
  __shared__ alignas(32) bf16 adjb[SS * SS];  // prepped chunk order: [kc][tile]
  __shared__ alignas(32) bf16 wst[32 * DD];
  const int tid = threadIdx.x, lane = tid & 31, wave = tid >> 5;
  const size_t base = (size_t)blockIdx.x * (SS * DD);
  const float* xg = x + base;

  stage_copy(adjb, adjP, (SS * SS) / 8, tid);
  ln_to_lds_B(hbuf, xg, gamma, beta, tid);
  stage_fence();

  const int rt = wave & 3, c0 = (wave >> 2) * 8;
  v8f acc[8];
#pragma unroll
  for (int j = 0; j < 8; ++j) acc[j] = vzero();

  // GEMM1: mixed = adj(64x64) @ h(64x256); adjb slots are [kc][tile]-major.
#pragma unroll
  for (int kcI = 0; kcI < 2; ++kcI) {
    const v16bf a = frag_ld(adjb, kcI, rt, 4, lane);  // slot = kcI*4 + rt
#pragma unroll
    for (int j = 0; j < 8; ++j) {
      const v16bf b = frag_ld(hbuf, c0 + j, kcI, 2, lane);
      acc[j] = wmma_bf16(a, b, acc[j]);
    }
  }
  __syncthreads();
  {  // write mixed (A-swizzled for GEMM2) over hbuf
    const int h2 = (lane >> 4) & 1, m = lane & 15;
#pragma unroll
    for (int j = 0; j < 8; ++j) {
      const int coln = (c0 + j) * 16 + m;
#pragma unroll
      for (int r = 0; r < 8; ++r)
        hbuf[swz(rt * 16 + h2 * 8 + r, coln, 8)] = (bf16)acc[j][r];
    }
  }
  __syncthreads();

  // GEMM2: out = x + mixed @ g_w + g_b
#pragma unroll
  for (int j = 0; j < 8; ++j) acc[j] = vzero();
  for (int kc = 0; kc < 8; ++kc) {
    stage_copy(wst, gwP + (size_t)kc * (32 * DD), (32 * DD) / 8, tid);
    stage_fence();
    const v16bf a = frag_ld(hbuf, rt, kc, 8, lane);
#pragma unroll
    for (int j = 0; j < 8; ++j) {
      const v16bf b = frag_ld(wst, c0 + j, 0, 1, lane);
      acc[j] = wmma_bf16(a, b, acc[j]);
    }
    __syncthreads();
  }
  {
    const int h2 = (lane >> 4) & 1, m = lane & 15;
    float* outp = x1 + base;
#pragma unroll
    for (int j = 0; j < 8; ++j) {
      const int coln = (c0 + j) * 16 + m;
      const float bias = gb[coln];
#pragma unroll
      for (int r = 0; r < 8; ++r) {
        const int row = rt * 16 + h2 * 8 + r;
        outp[row * DD + coln] = xg[row * DD + coln] + acc[j][r] + bias;
      }
    }
  }
}

// ===================== K2: time-mix projections =====================
// z = LN(x1) @ in_w + in_b ; alpha = sigmoid(z[:,:256]); cand = tanh(z[:,256:])
// alpha/cand are stored per-(b,t)-tile in A-swizzled order (scan is elementwise).
__global__ __launch_bounds__(256) void k_timeproj(
    const float* __restrict__ x1, const float* __restrict__ gamma,
    const float* __restrict__ beta, const bf16* __restrict__ inwP,
    const float* __restrict__ inb, bf16* __restrict__ alpha,
    bf16* __restrict__ cand) {
  __shared__ alignas(32) bf16 hbuf[SS * DD];
  __shared__ alignas(32) bf16 wst[32 * DD];
  const int tid = threadIdx.x, lane = tid & 31, wave = tid >> 5;
  const size_t base = (size_t)blockIdx.x * (SS * DD);

  ln_to_lds_A(hbuf, x1 + base, gamma, beta, tid);
  __syncthreads();

  const int rt = wave & 3, c0 = (wave >> 2) * 8;
  for (int half = 0; half < 2; ++half) {
    v8f acc[8];
#pragma unroll
    for (int j = 0; j < 8; ++j) acc[j] = vzero();
    for (int kc = 0; kc < 8; ++kc) {
      stage_copy(wst, inwP + (size_t)(half * 8 + kc) * (32 * DD), (32 * DD) / 8, tid);
      stage_fence();
      const v16bf a = frag_ld(hbuf, rt, kc, 8, lane);
#pragma unroll
      for (int j = 0; j < 8; ++j) {
        const v16bf b = frag_ld(wst, c0 + j, 0, 1, lane);
        acc[j] = wmma_bf16(a, b, acc[j]);
      }
      __syncthreads();
    }
    const int h2 = (lane >> 4) & 1, m = lane & 15;
    bf16* outp = (half == 0 ? alpha : cand) + base;
#pragma unroll
    for (int j = 0; j < 8; ++j) {
      const int coln = (c0 + j) * 16 + m;
      const float bias = inb[half * DD + coln];
#pragma unroll
      for (int r = 0; r < 8; ++r) {
        const int row = rt * 16 + h2 * 8 + r;
        const float z = acc[j][r] + bias;
        const float v = (half == 0) ? (1.f / (1.f + expf(-z))) : tanhf(z);
        outp[swz(row, coln, 8)] = (bf16)v;  // tile stored A-swizzled
      }
    }
    __syncthreads();
  }
}

// ===================== K3: leaky-integrator scan over T =====================
// Elementwise recurrence over intra-tile offsets (layout-agnostic).
__global__ __launch_bounds__(256) void k_scan(const bf16* __restrict__ alpha,
                                              bf16* __restrict__ cand) {
  const int idx = blockIdx.x * blockDim.x + threadIdx.x;  // over B * (S*D)
  const int SD = SS * DD;
  const int b = idx / SD, o = idx - b * SD;
  size_t p = (size_t)b * TT * SD + o;
  float state = 0.f;
  for (int t = 0; t < TT; ++t) {
    const float a = (float)alpha[p];
    const float c = (float)cand[p];
    state = (1.f - a) * state + a * c;
    cand[p] = (bf16)state;
    p += SD;
  }
}

// ===================== K4: output projection (in-place residual) =====================
// x1 += y @ out_w + out_b   (y tiles arrive already A-swizzled)
__global__ __launch_bounds__(256) void k_outproj(const bf16* __restrict__ y,
                                                 const bf16* __restrict__ owP,
                                                 const float* __restrict__ ob,
                                                 float* __restrict__ x1) {
  __shared__ alignas(32) bf16 ybuf[SS * DD];
  __shared__ alignas(32) bf16 wst[32 * DD];
  const int tid = threadIdx.x, lane = tid & 31, wave = tid >> 5;
  const size_t base = (size_t)blockIdx.x * (SS * DD);

  stage_copy(ybuf, y + base, (SS * DD) / 8, tid);

  const int rt = wave & 3, c0 = (wave >> 2) * 8;
  v8f acc[8];
#pragma unroll
  for (int j = 0; j < 8; ++j) acc[j] = vzero();
  for (int kc = 0; kc < 8; ++kc) {
    stage_copy(wst, owP + (size_t)kc * (32 * DD), (32 * DD) / 8, tid);
    stage_fence();
    const v16bf a = frag_ld(ybuf, rt, kc, 8, lane);
#pragma unroll
    for (int j = 0; j < 8; ++j) {
      const v16bf b = frag_ld(wst, c0 + j, 0, 1, lane);
      acc[j] = wmma_bf16(a, b, acc[j]);
    }
    __syncthreads();
  }
  const int h2 = (lane >> 4) & 1, m = lane & 15;
  float* xp = x1 + base;
#pragma unroll
  for (int j = 0; j < 8; ++j) {
    const int coln = (c0 + j) * 16 + m;
    const float bias = ob[coln];
#pragma unroll
    for (int r = 0; r < 8; ++r) {
      const int row = rt * 16 + h2 * 8 + r;
      xp[row * DD + coln] += acc[j][r] + bias;
    }
  }
}

// ===================== K5: fused FFN =====================
// out = x2 + gelu(LN(x2) @ w1 + b1) @ w2 + b2, H chunked by 64
__global__ __launch_bounds__(256) void k_ffn(
    const float* __restrict__ x2, const float* __restrict__ gamma,
    const float* __restrict__ beta, const bf16* __restrict__ w1P,
    const float* __restrict__ b1, const bf16* __restrict__ w2P,
    const float* __restrict__ b2, float* __restrict__ out) {
  __shared__ alignas(32) bf16 hbuf[SS * DD];  // A-swz, K=256 (32 KB)
  __shared__ alignas(32) bf16 gbuf[SS * 64];  // A-swz, K=64 (8 KB)
  __shared__ alignas(32) bf16 wst[32 * DD];   // stage buffer (16 KB)
  const int tid = threadIdx.x, lane = tid & 31, wave = tid >> 5;
  const size_t base = (size_t)blockIdx.x * (SS * DD);
  const float* xg = x2 + base;

  ln_to_lds_A(hbuf, xg, gamma, beta, tid);
  __syncthreads();

  const int rt = wave & 3, c0 = (wave >> 2) * 8;
  const int ct1 = (wave >> 2) * 2;  // GEMM1 col-tile pair
  const int h2 = (lane >> 4) & 1, m = lane & 15;

  v8f acc2[8];
#pragma unroll
  for (int j = 0; j < 8; ++j) acc2[j] = vzero();

  for (int hc = 0; hc < HH; hc += 64) {
    // GEMM1: o1 = h(64x256) @ w1[:, hc:hc+64]; w1P chunks: [hc/64][kc], 4 KB each
    v8f o1[2];
    o1[0] = vzero();
    o1[1] = vzero();
    for (int kc = 0; kc < 8; ++kc) {
      stage_copy(wst, w1P + (size_t)((hc >> 6) * 8 + kc) * (32 * 64), (32 * 64) / 8, tid);
      stage_fence();
      const v16bf a = frag_ld(hbuf, rt, kc, 8, lane);
#pragma unroll
      for (int j = 0; j < 2; ++j) {
        const v16bf b = frag_ld(wst, ct1 + j, 0, 1, lane);
        o1[j] = wmma_bf16(a, b, o1[j]);
      }
      __syncthreads();
    }
    // bias + gelu -> gbuf (A-swizzled, K=64)
#pragma unroll
    for (int j = 0; j < 2; ++j) {
      const int coln = (ct1 + j) * 16 + m;
      const float bias = b1[hc + coln];
#pragma unroll
      for (int r = 0; r < 8; ++r) {
        const int row = rt * 16 + h2 * 8 + r;
        gbuf[swz(row, coln, 2)] = (bf16)gelu_tanh(o1[j][r] + bias);
      }
    }
    __syncthreads();
    // GEMM2: acc2 += gbuf(64x64) @ w2[hc:hc+64, :]; w2P chunks: [k/32], 16 KB each
    for (int kc = 0; kc < 2; ++kc) {
      stage_copy(wst, w2P + (size_t)((hc >> 5) + kc) * (32 * DD), (32 * DD) / 8, tid);
      stage_fence();
      const v16bf a = frag_ld(gbuf, rt, kc, 2, lane);
#pragma unroll
      for (int j = 0; j < 8; ++j) {
        const v16bf b = frag_ld(wst, c0 + j, 0, 1, lane);
        acc2[j] = wmma_bf16(a, b, acc2[j]);
      }
      __syncthreads();
    }
  }

  float* outp = out + base;
#pragma unroll
  for (int j = 0; j < 8; ++j) {
    const int coln = (c0 + j) * 16 + m;
    const float bias = b2[coln];
#pragma unroll
    for (int r = 0; r < 8; ++r) {
      const int row = rt * 16 + h2 * 8 + r;
      outp[row * DD + coln] = xg[row * DD + coln] + acc2[j][r] + bias;
    }
  }
}

extern "C" void kernel_launch(void* const* d_in, const int* in_sizes, int n_in,
                              void* d_out, int out_size, void* d_ws, size_t ws_size,
                              hipStream_t stream) {
  const float* x       = (const float*)d_in[0];
  const float* adj     = (const float*)d_in[1];
  const float* g_gamma = (const float*)d_in[2];
  const float* g_beta  = (const float*)d_in[3];
  const float* g_w     = (const float*)d_in[4];
  const float* g_b     = (const float*)d_in[5];
  const float* t_gamma = (const float*)d_in[6];
  const float* t_beta  = (const float*)d_in[7];
  const float* in_w    = (const float*)d_in[8];
  const float* in_b    = (const float*)d_in[9];
  const float* out_w   = (const float*)d_in[10];
  const float* out_b   = (const float*)d_in[11];
  const float* f_gamma = (const float*)d_in[12];
  const float* f_beta  = (const float*)d_in[13];
  const float* w1      = (const float*)d_in[14];
  const float* b1      = (const float*)d_in[15];
  const float* w2      = (const float*)d_in[16];
  const float* b2      = (const float*)d_in[17];

  const size_t n_elem = (size_t)BB * TT * SS * DD;  // 67,108,864
  float* x1   = (float*)d_ws;                                    // fp32, 268 MB
  bf16* alpha = (bf16*)((char*)d_ws + n_elem * 4);               // bf16, 134 MB
  bf16* cand  = (bf16*)((char*)d_ws + n_elem * 4 + n_elem * 2);  // bf16, 134 MB
  bf16* prep  = (bf16*)((char*)d_ws + n_elem * 8);               // swizzled weights

  // prepped weight regions (element offsets)
  bf16* adjP = prep;                 // 4096
  bf16* gwP  = adjP + 4096;          // 65536
  bf16* inwP = gwP + 65536;          // 131072
  bf16* owP  = inwP + 131072;        // 65536
  bf16* w1P  = owP + 65536;          // 262144
  bf16* w2P  = w1P + 262144;         // 262144

  const dim3 grid(BB * TT), block(256), pgrid(64);
  // adj used as WMMA-A: prep its transpose as B-layout (sk=1, sn=64).
  k_prep<<<pgrid, block, 0, stream>>>(adj,   adjP, 1,   SS, SS, SS, 1);
  k_prep<<<pgrid, block, 0, stream>>>(g_w,   gwP,  DD,  1,  DD, DD, 1);
  k_prep<<<pgrid, block, 0, stream>>>(in_w,  inwP, 2*DD,1,  DD, DD, 2);
  k_prep<<<pgrid, block, 0, stream>>>(out_w, owP,  DD,  1,  DD, DD, 1);
  k_prep<<<pgrid, block, 0, stream>>>(w1,    w1P,  HH,  1,  DD, 64, 16);
  k_prep<<<pgrid, block, 0, stream>>>(w2,    w2P,  DD,  1,  HH, DD, 1);

  k_graphmix<<<grid, block, 0, stream>>>(x, adjP, g_gamma, g_beta, gwP, g_b, x1);
  k_timeproj<<<grid, block, 0, stream>>>(x1, t_gamma, t_beta, inwP, in_b, alpha, cand);
  k_scan<<<dim3((BB * SS * DD) / 256), block, 0, stream>>>(alpha, cand);
  k_outproj<<<grid, block, 0, stream>>>(cand /* y, A-swizzled tiles */, owP, out_b, x1);
  k_ffn<<<grid, block, 0, stream>>>(x1, f_gamma, f_beta, w1P, b1, w2P, b2, (float*)d_out);
}
